// BiLSTM_CRF_48679159332885
// MI455X (gfx1250) — matrix operand order repaired and checked by reading, not compiled, not run on hardware
//
#include <hip/hip_runtime.h>
#include <hip/hip_bf16.h>

typedef __attribute__((ext_vector_type(16))) __bf16 v16bf;
typedef __attribute__((ext_vector_type(8)))  float  v8f;
typedef __attribute__((ext_vector_type(4)))  unsigned u32x4;
typedef __attribute__((ext_vector_type(8)))  int      i32x8;
typedef __attribute__((ext_vector_type(4)))  int      i32x4;

#define SLEN 128
#define BATCH 32
#define HID   512
#define D0    1024
#define G4H   2048
#define ROWS  (SLEN*BATCH)   // 4096

// ---------------- WMMA helpers ----------------

__device__ __forceinline__ v8f wmma_bf16(v16bf a, v16bf b, v8f c) {
  return __builtin_amdgcn_wmma_f32_16x16x32_bf16(false, a, false, b, (short)0, c, false, false);
}

// A-matrix fragment (16x32, M x K), per ISA 7.12.2:
// lane<16 : row = lane,    K = k0+0..7  and k0+16..23
// lane>=16: row = lane-16, K = k0+8..15 and k0+24..31
__device__ __forceinline__ v16bf load_a_frag(const __bf16* base, int row0, int ld, int k0) {
  int lane = threadIdx.x & 31;
  const __bf16* p = base + (size_t)(row0 + (lane & 15)) * (size_t)ld
                         + (size_t)(k0 + ((lane >> 4) << 3));
  v16bf f;
  reinterpret_cast<uint4*>(&f)[0] = *reinterpret_cast<const uint4*>(p);
  reinterpret_cast<uint4*>(&f)[1] = *reinterpret_cast<const uint4*>(p + 16);
  return f;
}

// B-matrix fragment (32x16, K x N) loaded from row-major W[N, K] (so GEMM = X @ W^T):
// lane<16 : col = lane,    K = k0+0..7 and k0+16..23 ; lane>=16: col = lane-16, K = k0+8..15 and k0+24..31
__device__ __forceinline__ v16bf load_b_frag(const __bf16* base, int col0, int ld, int k0) {
  int lane = threadIdx.x & 31;
  const __bf16* p = base + (size_t)(col0 + (lane & 15)) * (size_t)ld
                         + (size_t)(k0 + ((lane >> 4) << 3));
  v16bf f;
  reinterpret_cast<uint4*>(&f)[0] = *reinterpret_cast<const uint4*>(p);
  reinterpret_cast<uint4*>(&f)[1] = *reinterpret_cast<const uint4*>(p + 16);
  return f;
}

__device__ __forceinline__ v8f zero8() {
  v8f z = {0.f,0.f,0.f,0.f,0.f,0.f,0.f,0.f};
  return z;
}

__device__ __forceinline__ float sigf(float x) { return 1.f / (1.f + __expf(-x)); }

// ---------------- elementwise prep ----------------

__global__ void k_f2bf(const float* __restrict__ x, __bf16* __restrict__ y, int n) {
  int i = blockIdx.x * blockDim.x + threadIdx.x;
  if (i < n) y[i] = (__bf16)x[i];
}

__global__ void k_build_emb(const float* __restrict__ src, const float* __restrict__ bert,
                            __bf16* __restrict__ emb) {
  int i = blockIdx.x * blockDim.x + threadIdx.x;     // over 4096*1024
  if (i >= ROWS * D0) return;
  int row = i >> 10, k = i & 1023;
  int b = row & 31;
  float v = (k < 256) ? src[row * 256 + k] : bert[b * 768 + (k - 256)];
  emb[i] = (__bf16)v;
}

__global__ void k_bias_comb(const float* __restrict__ bih, const float* __restrict__ bhh,
                            float* __restrict__ bc, int n) {
  int i = blockIdx.x * blockDim.x + threadIdx.x;
  if (i < n) bc[i] = bih[i] + bhh[i];
}

__global__ void k_zero_out(float* o) { if (threadIdx.x == 0) o[0] = 0.f; }

// ---------------- xW = X @ W_ih^T + bias  (both dirs of one layer) ----------------
// X: [4096,1024] bf16, W: [2][2048,1024] bf16, bias: [2][2048] f32, xW: [2][4096,2048] f32
// Block tile: 32(M) x 512(N) of one direction; 8 waves of 32x64.
// The shared 32x1024 bf16 X tile (64 KB) is DMA'd into LDS by the Tensor Data Mover,
// then A-fragments are read from LDS while B streams from L2.
__global__ void k_gemm_xw(const __bf16* __restrict__ X, const __bf16* __restrict__ W,
                          const float* __restrict__ bias, float* __restrict__ xW) {
  int bid = blockIdx.x;            // 2 * 128 * 4 = 1024 blocks
  int d   = bid >> 9;
  int rem = bid & 511;
  int mt  = rem >> 2;              // 0..127 -> rows mt*32
  int nb  = rem & 3;               // 0..3   -> cols nb*512
  const __bf16* Wd  = W  + (size_t)d * G4H * D0;
  float*        xWd = xW + (size_t)d * ROWS * G4H;
  const float*  bd  = bias + d * G4H;
  int m0 = mt * 32;

  __shared__ __align__(16) __bf16 xtile[32 * D0];    // 64 KB

  // TDM: one wave issues the descriptor; tile = rows [m0, m0+32) x [0, 1024)
  if ((threadIdx.x >> 5) == 0) {
    unsigned lds = (unsigned)(uintptr_t)(&xtile[0]);
    unsigned long long ga = (unsigned long long)(uintptr_t)(X + (size_t)m0 * D0);
    u32x4 g0;
    g0[0] = 1u;                                             // count=1, user mode
    g0[1] = lds;                                            // lds_addr (bytes)
    g0[2] = (unsigned)(ga & 0xffffffffu);                   // global_addr[31:0]
    g0[3] = (unsigned)((ga >> 32) & 0x01ffffffu) | (2u << 30); // addr[56:32] | type=2
    i32x8 g1;
    g1[0] = (int)(1u << 16);                                // data_size=1 -> 2 bytes
    g1[1] = (int)((D0 & 0xffffu) << 16);                    // tensor_dim0[15:0]
    g1[2] = (int)(((unsigned)D0 >> 16) | ((ROWS & 0xffffu) << 16)); // dim0 hi | dim1 lo
    g1[3] = (int)(((unsigned)ROWS >> 16) | ((unsigned)D0 << 16));   // dim1 hi | tile_dim0
    g1[4] = (int)32;                                        // tile_dim1=32, tile_dim2=0
    g1[5] = (int)D0;                                        // tensor_dim0_stride[31:0]
    g1[6] = 0;                                              // stride0 hi | stride1 lo
    g1[7] = 0;                                              // stride1 hi
    i32x4 g2 = {0, 0, 0, 0};
    i32x4 g3 = {0, 0, 0, 0};
    i32x8 g4 = {0, 0, 0, 0, 0, 0, 0, 0};
    __builtin_amdgcn_tensor_load_to_lds(g0, g1, g2, g3, g4, 0);
    __builtin_amdgcn_s_wait_tensorcnt(0);
  }
  __syncthreads();

  int wv = threadIdx.x >> 5;       // 0..7
  int n0 = nb * 512 + wv * 64;

  v8f acc[2][4];
  #pragma unroll
  for (int i = 0; i < 2; ++i)
    #pragma unroll
    for (int j = 0; j < 4; ++j) acc[i][j] = zero8();

  for (int k0 = 0; k0 < D0; k0 += 32) {
    v16bf a0 = load_a_frag(xtile, 0,  D0, k0);   // LDS
    v16bf a1 = load_a_frag(xtile, 16, D0, k0);   // LDS
    #pragma unroll
    for (int j = 0; j < 4; ++j) {
      v16bf b = load_b_frag(Wd, n0 + j * 16, D0, k0);
      acc[0][j] = wmma_bf16(a0, b, acc[0][j]);
      acc[1][j] = wmma_bf16(a1, b, acc[1][j]);
    }
  }

  int lane = threadIdx.x & 31;
  #pragma unroll
  for (int mi = 0; mi < 2; ++mi) {
    float* dstrow = xWd + (size_t)(m0 + mi * 16 + ((lane >> 4) << 3)) * G4H;
    #pragma unroll
    for (int j = 0; j < 4; ++j) {
      int col = n0 + j * 16 + (lane & 15);
      float bb = bd[col];
      #pragma unroll
      for (int r = 0; r < 8; ++r)
        dstrow[(size_t)r * G4H + col] = acc[mi][j][r] + bb;
    }
  }
}

// ---------------- recurrent LSTM scan (one block per direction) ----------------
// xW: [2][4096,2048] f32 (gate pre-activations from input side, biases included)
// whh: [2][2048,512] bf16, out: [4096, 1024] bf16 (dir d writes cols d*512..+512)
__global__ void k_lstm_rec(const float* __restrict__ xW, const __bf16* __restrict__ whh,
                           __bf16* __restrict__ out) {
  int d = blockIdx.x;
  const float*  xWd = xW  + (size_t)d * ROWS * G4H;
  const __bf16* Wd  = whh + (size_t)d * G4H * HID;

  __shared__ __align__(16) __bf16 hbuf[BATCH * HID];   // 32 KB
  for (int i = threadIdx.x; i < BATCH * HID; i += blockDim.x) hbuf[i] = (__bf16)0.f;
  __syncthreads();

  int w    = threadIdx.x >> 5;
  int lane = threadIdx.x & 31;
  int m    = w & 1;        // batch-row tile: rows m*16..+16
  int ug   = w >> 1;       // hidden-unit group: u-tiles 2*ug, 2*ug+1

  v8f cst[2];              // cell state, per u-tile, register-resident
  cst[0] = zero8(); cst[1] = zero8();

  for (int t = 0; t < SLEN; ++t) {
    int tt = d ? (SLEN - 1 - t) : t;

    // prefetch next step's gate pre-activations (latency-bound scan)
    if (t + 1 < SLEN) {
      int tn = d ? (SLEN - 2 - t) : (t + 1);
      const float* pf = xWd + (size_t)(tn * BATCH + m * 16 + ((lane >> 4) << 3)) * G4H
                            + (ug * 2) * 16 + (lane & 15);
      __builtin_prefetch(pf, 0, 1);
    }

    v8f acc[2][4];         // [u-tile][gate i,f,g,o]
    #pragma unroll
    for (int i = 0; i < 2; ++i)
      #pragma unroll
      for (int j = 0; j < 4; ++j) acc[i][j] = zero8();

    for (int k0 = 0; k0 < HID; k0 += 32) {
      v16bf a = load_a_frag(hbuf, m * 16, HID, k0);   // h from LDS
      #pragma unroll
      for (int ui = 0; ui < 2; ++ui) {
        int urow = (ug * 2 + ui) * 16;
        #pragma unroll
        for (int g = 0; g < 4; ++g) {
          v16bf b = load_b_frag(Wd, g * HID + urow, HID, k0);
          acc[ui][g] = wmma_bf16(a, b, acc[ui][g]);
        }
      }
    }
    __syncthreads();       // everyone done reading old h

    const float* gx = xWd + (size_t)(tt * BATCH + m * 16 + ((lane >> 4) << 3)) * G4H;
    #pragma unroll
    for (int ui = 0; ui < 2; ++ui) {
      int ucol = (ug * 2 + ui) * 16 + (lane & 15);
      #pragma unroll
      for (int r = 0; r < 8; ++r) {
        float gi = sigf (acc[ui][0][r] + gx[(size_t)r * G4H + 0 * HID + ucol]);
        float gf = sigf (acc[ui][1][r] + gx[(size_t)r * G4H + 1 * HID + ucol]);
        float gg = tanhf(acc[ui][2][r] + gx[(size_t)r * G4H + 2 * HID + ucol]);
        float go = sigf (acc[ui][3][r] + gx[(size_t)r * G4H + 3 * HID + ucol]);
        float c  = gf * cst[ui][r] + gi * gg;
        cst[ui][r] = c;
        float h  = go * tanhf(c);
        int   M  = m * 16 + ((lane >> 4) << 3) + r;
        hbuf[M * HID + ucol] = (__bf16)h;
        out[(size_t)(tt * BATCH + M) * (2 * HID) + d * HID + ucol] = (__bf16)h;
      }
    }
    __syncthreads();       // new h visible before next step's reads
  }
}

// ---------------- heads: feat = out @ head_w^T + b (pos: N=32, morph: N=64) ----------------
__global__ void k_heads(const __bf16* __restrict__ X,
                        const __bf16* __restrict__ wp, const float* __restrict__ bp,
                        const __bf16* __restrict__ wm, const float* __restrict__ bm,
                        float* __restrict__ fp, float* __restrict__ fm) {
  int wid = blockIdx.x * (blockDim.x >> 5) + (threadIdx.x >> 5);
  if (wid >= 256 * 6) return;
  int mt = wid / 6, nj = wid % 6;
  const __bf16* W; const float* bias; float* dst; int col0, ldc;
  if (nj < 2) { W = wp; bias = bp; dst = fp; col0 = nj * 16;       ldc = 32; }
  else        { W = wm; bias = bm; dst = fm; col0 = (nj - 2) * 16; ldc = 64; }

  v8f acc = zero8();
  for (int k0 = 0; k0 < 2 * HID; k0 += 32) {
    v16bf a = load_a_frag(X, mt * 16, 2 * HID, k0);
    v16bf b = load_b_frag(W, col0,    2 * HID, k0);
    acc = wmma_bf16(a, b, acc);
  }
  int lane = threadIdx.x & 31;
  int col  = col0 + (lane & 15);
  float bb = bias[col];
  float* p = dst + (size_t)(mt * 16 + ((lane >> 4) << 3)) * ldc + col;
  #pragma unroll
  for (int r = 0; r < 8; ++r) p[(size_t)r * ldc] = acc[r] + bb;
}

// ---------------- CRF log-likelihood (forward algorithm), one block per task ----------------
__global__ void k_crf(const float* __restrict__ fp, const float* __restrict__ fm,
                      const int* __restrict__ lp, const int* __restrict__ lm,
                      const float* __restrict__ sp, const float* __restrict__ ep, const float* __restrict__ tp,
                      const float* __restrict__ sm, const float* __restrict__ emv, const float* __restrict__ tm,
                      float* __restrict__ outv) {
  int task = blockIdx.x;
  int K = task ? 64 : 32;
  const float* em   = task ? fm  : fp;
  const int*   tags = task ? lm  : lp;
  const float* st   = task ? sm  : sp;
  const float* en   = task ? emv : ep;
  const float* tr   = task ? tm  : tp;

  __shared__ float trs[64 * 64];
  __shared__ float alpha[2][BATCH * 64];
  __shared__ float red[BATCH];

  for (int i = threadIdx.x; i < K * K; i += blockDim.x) trs[i] = tr[i];
  for (int i = threadIdx.x; i < BATCH * K; i += blockDim.x) {
    int b = i / K, j = i % K;
    alpha[0][i] = st[j] + em[(size_t)b * K + j];
  }
  __syncthreads();

  int cur = 0;
  for (int t = 1; t < SLEN; ++t) {
    for (int i = threadIdx.x; i < BATCH * K; i += blockDim.x) {
      int b = i / K, j = i % K;
      const float* ab = &alpha[cur][b * K];
      float mx = -INFINITY;
      for (int q = 0; q < K; ++q) mx = fmaxf(mx, ab[q] + trs[q * K + j]);
      float s = 0.f;
      for (int q = 0; q < K; ++q) s += __expf(ab[q] + trs[q * K + j] - mx);
      alpha[cur ^ 1][i] = mx + __logf(s) + em[(size_t)(t * BATCH + b) * K + j];
    }
    __syncthreads();
    cur ^= 1;
  }

  if (threadIdx.x < BATCH) {
    int b = threadIdx.x;
    const float* ab = &alpha[cur][b * K];
    float mx = -INFINITY;
    for (int j = 0; j < K; ++j) mx = fmaxf(mx, ab[j] + en[j]);
    float s = 0.f;
    for (int j = 0; j < K; ++j) s += __expf(ab[j] + en[j] - mx);
    float den = mx + __logf(s);

    int prev = tags[b];
    float num = st[prev] + em[(size_t)b * K + prev];
    for (int t = 1; t < SLEN; ++t) {
      int tg = tags[t * BATCH + b];
      num += trs[prev * K + tg] + em[(size_t)(t * BATCH + b) * K + tg];
      prev = tg;
    }
    num += en[prev];
    red[b] = num - den;
  }
  __syncthreads();
  if (threadIdx.x == 0) {
    float s = 0.f;
    for (int b = 0; b < BATCH; ++b) s += red[b];
    atomicAdd(outv, -s);
  }
}

// ---------------- host launcher ----------------

extern "C" void kernel_launch(void* const* d_in, const int* in_sizes, int n_in,
                              void* d_out, int out_size, void* d_ws, size_t ws_size,
                              hipStream_t stream) {
  const float* src  = (const float*)d_in[0];
  const float* bert = (const float*)d_in[1];
  const int*   lp   = (const int*)d_in[2];
  const int*   lm   = (const int*)d_in[3];
  const float* wih  = (const float*)d_in[4];
  const float* whh  = (const float*)d_in[5];
  const float* bih  = (const float*)d_in[6];
  const float* bhh  = (const float*)d_in[7];
  const float* hwp  = (const float*)d_in[8];
  const float* hbp  = (const float*)d_in[9];
  const float* hwm  = (const float*)d_in[10];
  const float* hbm  = (const float*)d_in[11];
  const float* sp   = (const float*)d_in[12];
  const float* ep   = (const float*)d_in[13];
  const float* tp   = (const float*)d_in[14];
  const float* sm   = (const float*)d_in[15];
  const float* emv  = (const float*)d_in[16];
  const float* tm   = (const float*)d_in[17];

  char* ws = (char*)d_ws;
  size_t off = 0;
  auto alloc = [&](size_t bytes) { void* p = ws + off; off += (bytes + 255) & ~(size_t)255; return p; };

  __bf16* bufA   = (__bf16*)alloc((size_t)ROWS * D0 * 2);            // emb / layer-1 out
  __bf16* bufB   = (__bf16*)alloc((size_t)ROWS * D0 * 2);            // layer-0 out
  __bf16* wihb   = (__bf16*)alloc((size_t)4 * G4H * D0 * 2);
  __bf16* whhb   = (__bf16*)alloc((size_t)4 * G4H * HID * 2);
  __bf16* wpb    = (__bf16*)alloc((size_t)32 * D0 * 2);
  __bf16* wmb    = (__bf16*)alloc((size_t)64 * D0 * 2);
  float*  biasc  = (float*)alloc((size_t)4 * G4H * 4);
  float*  xW     = (float*)alloc((size_t)2 * ROWS * G4H * 4);
  float*  featp  = (float*)alloc((size_t)ROWS * 32 * 4);
  float*  featm  = (float*)alloc((size_t)ROWS * 64 * 4);

  // prep
  {
    int n = ROWS * D0;
    k_build_emb<<<(n + 255) / 256, 256, 0, stream>>>(src, bert, bufA);
  }
  { int n = 4 * G4H * D0;  k_f2bf<<<(n + 255) / 256, 256, 0, stream>>>(wih, wihb, n); }
  { int n = 4 * G4H * HID; k_f2bf<<<(n + 255) / 256, 256, 0, stream>>>(whh, whhb, n); }
  { int n = 32 * D0;       k_f2bf<<<(n + 255) / 256, 256, 0, stream>>>(hwp, wpb, n); }
  { int n = 64 * D0;       k_f2bf<<<(n + 255) / 256, 256, 0, stream>>>(hwm, wmb, n); }
  { int n = 4 * G4H;       k_bias_comb<<<(n + 255) / 256, 256, 0, stream>>>(bih, bhh, biasc, n); }
  k_zero_out<<<1, 1, 0, stream>>>((float*)d_out);

  // two BiLSTM layers
  for (int l = 0; l < 2; ++l) {
    const __bf16* in  = l ? bufB : bufA;
    __bf16*       ou  = l ? bufA : bufB;
    k_gemm_xw<<<1024, 256, 0, stream>>>(in, wihb + (size_t)l * 2 * G4H * D0,
                                        biasc + (size_t)l * 2 * G4H, xW);
    k_lstm_rec<<<2, 1024, 0, stream>>>(xW, whhb + (size_t)l * 2 * G4H * HID, ou);
  }

  // heads + CRF
  k_heads<<<192, 256, 0, stream>>>(bufA, wpb, hbp, wmb, hbm, featp, featm);
  k_crf<<<2, 1024, 0, stream>>>(featp, featm, lp, lm, sp, ep, tp, sm, emv, tm, (float*)d_out);
}
